// FeedForwardQuantum_65481071396818
// MI455X (gfx1250) — matrix operand order repaired and checked by reading, not compiled, not run on hardware
//
#include <hip/hip_runtime.h>

typedef _Float16 v16h __attribute__((ext_vector_type(16)));
typedef _Float16 v8h  __attribute__((ext_vector_type(8)));
typedef float    v8f  __attribute__((ext_vector_type(8)));

#define T_TOK 16384   // 4 * 4096 tokens
#define E_DIM 1024
#define F_DIM 4096
#define NQ    10
#define BM    128     // tokens per block
#define BN    128     // output features per block
#define BK    32      // K-chunk (= wmma depth)

__global__ __launch_bounds__(256) void ffq_fused_kernel(
    const float* __restrict__ x, const float* __restrict__ theta,
    const float* __restrict__ W1, const float* __restrict__ b1,
    const float* __restrict__ W2, const float* __restrict__ b2,
    float* __restrict__ out)
{
    __shared__ _Float16 Ws[BN][BK];    // 8 KB : W2 tile f16, n-major (k contiguous)
    __shared__ _Float16 Hs[BM][BK];    // 8 KB : h tile f16, m-major (k contiguous)
    __shared__ _Float16 qfh[BM][BK];   // 8 KB : augmented features f16 (j contiguous, j=10 is bias-1)
    __shared__ _Float16 W1h[BK][BK];   // 2 KB : augmented W1 chunk f16 (j contiguous, j=10 is b1)

    const int t    = threadIdx.x;
    const int lane = t & 31;
    const int wave = t >> 5;
    const int wm   = wave >> 2;            // 0..1 : 64-wide n strip (main GEMM M-dim)
    const int wn2  = wave & 3;             // 0..3 : 32-wide m strip (main GEMM N-dim)
    const int m0   = blockIdx.y * BM;
    const int n0   = blockIdx.x * BN;
    const int lr   = lane & 15;            // row/col within 16
    const int lh   = lane >> 4;            // half-wave select

    // ---- stage augmented token features once (threads 0..127, one token each)
    // qfaug[m][j] = cos(x[m][j]+theta[j]) for j<10, 1.0 at j==10 (bias), 0 elsewhere
    if (t < BM) {
        const float* xr = x + (size_t)(m0 + t) * E_DIM;
        float cv[NQ];
        #pragma unroll
        for (int j = 0; j < NQ; ++j) cv[j] = cosf(xr[j] + theta[j]);
        v8h q0, q1, z = {};
        #pragma unroll
        for (int j = 0; j < 8; ++j) q0[j] = (_Float16)cv[j];
        q1 = z;
        q1[0] = (_Float16)cv[8];
        q1[1] = (_Float16)cv[9];
        q1[2] = (_Float16)1.0f;            // bias feature
        v8h* qr = (v8h*)&qfh[t][0];
        qr[0] = q0; qr[1] = q1; qr[2] = z; qr[3] = z;
    }
    __syncthreads();

    // ---- loop-invariant B-fragment of the h-gen GEMM: qf columns for m-tile = wave
    // B1[j][m]: lane holds col m = wave*16+lr, j = lh*8+0..7 and 16+lh*8+0..7
    v16h bq;
    {
        const _Float16* p = &qfh[wave * 16 + lr][lh * 8];
        ((float4*)&bq)[0] = *(const float4*)p;
        ((float4*)&bq)[1] = *(const float4*)(p + 16);
    }

    v8f c[4][2];
    #pragma unroll
    for (int nt = 0; nt < 4; ++nt)
        #pragma unroll
        for (int mt = 0; mt < 2; ++mt) { v8f z = {}; c[nt][mt] = z; }

    for (int kc = 0; kc < F_DIM; kc += BK) {
        __syncthreads();   // previous iteration's LDS fragment reads complete

        // ---- stage augmented W1 chunk (uniform t<32 path, aligned float2 loads)
        if (t < BK) {
            const float* wr = W1 + (size_t)(kc + t) * NQ;     // 40B row stride, 8B aligned
            float2 a0 = *(const float2*)(wr + 0);
            float2 a1 = *(const float2*)(wr + 2);
            float2 a2 = *(const float2*)(wr + 4);
            float2 a3 = *(const float2*)(wr + 6);
            float2 a4 = *(const float2*)(wr + 8);
            float  bb = b1[kc + t];
            v8h h0, h1, z = {};
            h0[0]=(_Float16)a0.x; h0[1]=(_Float16)a0.y; h0[2]=(_Float16)a1.x; h0[3]=(_Float16)a1.y;
            h0[4]=(_Float16)a2.x; h0[5]=(_Float16)a2.y; h0[6]=(_Float16)a3.x; h0[7]=(_Float16)a3.y;
            h1 = z;
            h1[0]=(_Float16)a4.x; h1[1]=(_Float16)a4.y; h1[2]=(_Float16)bb;   // b1 rides j==10
            v8h* wrow = (v8h*)&W1h[t][0];
            wrow[0] = h0; wrow[1] = h1; wrow[2] = z; wrow[3] = z;
        }

        // ---- stage W2 tile [128n x 32k] f32 -> f16
        #pragma unroll
        for (int i = 0; i < 2; ++i) {
            int idx = t + i * 256;
            int n   = idx >> 2;
            int kq  = (idx & 3) * 8;
            const float* src = W2 + (size_t)(n0 + n) * F_DIM + kc + kq;
            float4 lo = ((const float4*)src)[0];
            float4 hi = ((const float4*)src)[1];
            v8h hb;
            hb[0]=(_Float16)lo.x; hb[1]=(_Float16)lo.y; hb[2]=(_Float16)lo.z; hb[3]=(_Float16)lo.w;
            hb[4]=(_Float16)hi.x; hb[5]=(_Float16)hi.y; hb[6]=(_Float16)hi.z; hb[7]=(_Float16)hi.w;
            *(v8h*)&Ws[n][kq] = hb;
        }
        __syncthreads();   // W1h / Ws visible

        // ---- h-generation via WMMA: hT[k][m] tile for m-tile = wave, both k-halves
        // D1 = W1aug(16k x 32j) x qfaug(32j x 16m); bias folded via j==10
        #pragma unroll
        for (int kt = 0; kt < 2; ++kt) {
            v16h a1f;
            const _Float16* ap = &W1h[kt * 16 + lr][lh * 8];
            ((float4*)&a1f)[0] = *(const float4*)ap;
            ((float4*)&a1f)[1] = *(const float4*)(ap + 16);
            v8f zc = {};
            v8f d = __builtin_amdgcn_wmma_f32_16x16x32_f16(
                false, a1f, false, bq, (short)0, zc, false, false);
            // relu + cvt; lane holds col m, rows k = kt*16 + 8*lh + r  (contiguous!)
            v8h hh;
            #pragma unroll
            for (int r = 0; r < 8; ++r)
                hh[r] = (_Float16)fmaxf(d[r], 0.f);
            *(v8h*)&Hs[wave * 16 + lr][kt * 16 + 8 * lh] = hh;   // one b128 store
        }
        __syncthreads();   // Hs visible

        // ---- main GEMM (transposed): D2[n][m] += W2tile(16n x 32k) x hT(32k x 16m)
        v16h bh[2];
        #pragma unroll
        for (int mt = 0; mt < 2; ++mt) {
            const _Float16* p = &Hs[wn2 * 32 + mt * 16 + lr][lh * 8];
            ((float4*)&bh[mt])[0] = *(const float4*)p;
            ((float4*)&bh[mt])[1] = *(const float4*)(p + 16);
        }
        #pragma unroll
        for (int nt = 0; nt < 4; ++nt) {
            v16h aw;
            const _Float16* ap = &Ws[wm * 64 + nt * 16 + lr][lh * 8];
            ((float4*)&aw)[0] = *(const float4*)ap;
            ((float4*)&aw)[1] = *(const float4*)(ap + 16);
            #pragma unroll
            for (int mt = 0; mt < 2; ++mt)
                c[nt][mt] = __builtin_amdgcn_wmma_f32_16x16x32_f16(
                    false, aw, false, bh[mt], (short)0, c[nt][mt], false, false);
        }
    }

    // ---- epilogue: D2 lane holds col m, rows n = nb + r (contiguous)
    //      -> two coalesced b128 stores per tile, bias b2[n] added from float4 loads
    #pragma unroll
    for (int nt = 0; nt < 4; ++nt)
        #pragma unroll
        for (int mt = 0; mt < 2; ++mt) {
            int m  = m0 + wn2 * 32 + mt * 16 + lr;
            int nb = n0 + wm * 64 + nt * 16 + 8 * lh;
            const float4* bp = (const float4*)(b2 + nb);
            float4 e0 = bp[0], e1 = bp[1];
            v8f d = c[nt][mt];
            float4 o0, o1;
            o0.x = d[0] + e0.x; o0.y = d[1] + e0.y; o0.z = d[2] + e0.z; o0.w = d[3] + e0.w;
            o1.x = d[4] + e1.x; o1.y = d[5] + e1.y; o1.z = d[6] + e1.z; o1.w = d[7] + e1.w;
            float4* op = (float4*)(out + (size_t)m * E_DIM + nb);
            op[0] = o0; op[1] = o1;
        }
}

extern "C" void kernel_launch(void* const* d_in, const int* in_sizes, int n_in,
                              void* d_out, int out_size, void* d_ws, size_t ws_size,
                              hipStream_t stream) {
    (void)in_sizes; (void)n_in; (void)out_size; (void)d_ws; (void)ws_size;
    const float* x     = (const float*)d_in[0];
    const float* theta = (const float*)d_in[1];
    const float* W1    = (const float*)d_in[2];
    const float* b1    = (const float*)d_in[3];
    const float* W2    = (const float*)d_in[4];
    const float* b2    = (const float*)d_in[5];
    float* out = (float*)d_out;

    dim3 grid(E_DIM / BN, T_TOK / BM);   // (8, 128) workgroups
    ffq_fused_kernel<<<grid, dim3(256), 0, stream>>>(x, theta, W1, b1, W2, b2, out);
}
